// HyenaOperator_37374805409916
// MI455X (gfx1250) — matrix operand order repaired and checked by reading, not compiled, not run on hardware
//
#include <hip/hip_runtime.h>
#include <math.h>

typedef __attribute__((ext_vector_type(16))) _Float16 v16h;
typedef __attribute__((ext_vector_type(8)))  float    v8f;
typedef __attribute__((ext_vector_type(4)))  unsigned int v4u;
typedef __attribute__((ext_vector_type(8)))  int      v8i;
typedef __attribute__((ext_vector_type(4)))  int      v4i;

union FragH { v16h v; uint4 q[2]; };

#define TC 64   // time chunk for long conv
#define DT 64   // channel tile for long conv

// ---------------- TDM: 2D f16 tile (tile_h rows x tile_w elems, row stride) -> LDS ----------------
// D# per CDNA5 ISA. Groups 2/3 zero (2D tensor), 6-arg builtin form.
__device__ __forceinline__ void tdm_load_2d_f16(const _Float16* gsrc, unsigned lds_off,
                                                unsigned tile_w, unsigned tile_h,
                                                unsigned stride_elems) {
  unsigned long long ga = (unsigned long long)(uintptr_t)gsrc;
  v4u g0 = { 1u,                                        // count=1, user descriptor
             lds_off,                                   // lds_addr (bytes)
             (unsigned)(ga & 0xffffffffu),              // global_addr[31:0]
             (unsigned)((ga >> 32) & 0x01ffffffu) | (2u << 30) };  // ga[56:32] | type=2
  v8i g1 = { (int)(1u << 16),        // workgroup_mask=0 (not in cluster), data_size=1 (2B)
             (int)(tile_w << 16),    // atomic_barrier_addr=0 | tensor_dim0[15:0]=tile_w
             (int)(tile_h << 16),    // tensor_dim0[31:16]=0 | tensor_dim1[15:0]=tile_h
             (int)(tile_w << 16),    // tensor_dim1[31:16]=0 | tile_dim0=tile_w
             (int)tile_h,            // tile_dim1=tile_h | tile_dim2=0
             (int)stride_elems,      // tensor_dim0_stride[31:0] (elements)
             0,                      // stride hi | tensor_dim1_stride lo (unused, 2D)
             0 };
  v4i z4 = { 0, 0, 0, 0 };
  v8i z8 = { 0, 0, 0, 0, 0, 0, 0, 0 };
  __builtin_amdgcn_tensor_load_to_lds(g0, g1, z4, z4, z8, 0);
}

// ---------------- f32 -> f16 elementwise ----------------
__global__ void cvt_f16(const float* __restrict__ s, _Float16* __restrict__ d, int n) {
  int i = blockIdx.x * 256 + threadIdx.x;
  if (i < n) d[i] = (_Float16)s[i];
}

// ---------------- f32 [K][N] -> f16 [N][K] (weight transpose) ----------------
__global__ void cvt_transpose(const float* __restrict__ w, _Float16* __restrict__ wt,
                              int K, int N) {
  int i = blockIdx.x * 256 + threadIdx.x;
  if (i < N * K) {
    int n = i / K, k = i % K;
    wt[i] = (_Float16)w[(size_t)k * N + n];
  }
}

// ---------------- implicit filter h[t][d] ----------------
__device__ __forceinline__ float gelu_exact(float x) {
  return 0.5f * x * (1.0f + erff(x * 0.70710678118654752f));
}

__global__ __launch_bounds__(256) void hyena_filter(
    const float* __restrict__ freqs, const float* __restrict__ w1, const float* __restrict__ b1,
    const float* __restrict__ w2, const float* __restrict__ b2,
    const float* __restrict__ w3, const float* __restrict__ b3,
    const float* __restrict__ decay, float* __restrict__ h, int Lc, int Dc) {
  __shared__ float enc[8];
  __shared__ float h1[64];
  __shared__ float h2[64];
  const int t   = blockIdx.x;
  const int tid = threadIdx.x;
  const float tl = (float)t / (float)(Lc - 1);
  if (tid < 7) {
    float v;
    if      (tid < 3) v = sinf(6.28318530717958647f * freqs[tid] * tl);
    else if (tid < 6) v = cosf(6.28318530717958647f * freqs[tid - 3] * tl);
    else              v = tl;
    enc[tid] = v;
  }
  __syncthreads();
  if (tid < 64) {
    float a = b1[tid];
    #pragma unroll
    for (int j = 0; j < 7; j++) a += enc[j] * w1[j * 64 + tid];
    h1[tid] = gelu_exact(a);
  }
  __syncthreads();
  if (tid < 64) {
    float a = b2[tid];
    for (int j = 0; j < 64; j++) a += h1[j] * w2[j * 64 + tid];
    h2[tid] = gelu_exact(a);
  }
  __syncthreads();
  const float wexp = tl * (float)Lc;
  for (int r = 0; r < 4; r++) {
    int dcol = r * 256 + tid;
    float a = b3[dcol];
    for (int j = 0; j < 64; j++) a += h2[j] * w3[j * Dc + dcol];
    h[(size_t)t * Dc + dcol] = a * expf(-fabsf(decay[dcol]) * wexp);
  }
}

// ---------------- causal depthwise long conv (chunked Toeplitz) ----------------
__global__ __launch_bounds__(256) void hyena_longconv(
    const float* __restrict__ vin, int vstride,
    const float* __restrict__ h, float* __restrict__ out, int Lc, int Dc) {
  __shared__ __align__(16) float vs[TC * DT];
  __shared__ __align__(16) float hs[(2 * TC - 1) * DT];
  const int tid = threadIdx.x;
  const int d   = tid & 63;
  const int tg  = tid >> 6;                 // 0..3
  const int dt0 = blockIdx.x * DT;
  const int I   = blockIdx.y;               // output time chunk
  const int b   = blockIdx.z;
  const int tbase = tg * 16;

  float acc[16];
  #pragma unroll
  for (int i = 0; i < 16; i++) acc[i] = 0.0f;

  // diagonal chunk (J == I): triangular
  for (int idx = tid; idx < TC * DT; idx += 256) {
    int s = idx >> 6, dd = idx & 63;
    vs[idx] = vin[(size_t)(b * Lc + I * TC + s) * vstride + dt0 + dd];
    hs[idx] = h[(size_t)(idx >> 6) * Dc + dt0 + dd];
  }
  __syncthreads();
  for (int s = 0; s < TC; s++) {
    float vv = vs[s * DT + d];
    #pragma unroll
    for (int ti = 0; ti < 16; ti++) {
      int t = tbase + ti;
      if (s <= t) acc[ti] += vv * hs[(t - s) * DT + d];
    }
  }
  __syncthreads();

  for (int J = I - 1; J >= 0; J--) {
    if (J > 0)  // warm next v chunk (global_prefetch_b8)
      __builtin_prefetch(&vin[(size_t)(b * Lc + (J - 1) * TC) * vstride + dt0 + d], 0, 0);
    const int off = I - J;
    const int hbase = off * TC - (TC - 1);   // >= 1
    for (int idx = tid; idx < TC * DT; idx += 256) {
      int s = idx >> 6, dd = idx & 63;
      vs[idx] = vin[(size_t)(b * Lc + J * TC + s) * vstride + dt0 + dd];
    }
    for (int idx = tid; idx < (2 * TC - 1) * DT; idx += 256) {
      int o = idx >> 6, dd = idx & 63;
      hs[idx] = h[(size_t)(hbase + o) * Dc + dt0 + dd];
    }
    __syncthreads();
    for (int s = 0; s < TC; s++) {
      float vv = vs[s * DT + d];
      int rb = (TC - 1 + tbase - s) * DT + d;
      #pragma unroll
      for (int ti = 0; ti < 16; ti++) acc[ti] += vv * hs[rb + ti * DT];
    }
    __syncthreads();
  }

  #pragma unroll
  for (int ti = 0; ti < 16; ti++) {
    int t = tbase + ti;
    out[(size_t)(b * Lc + I * TC + t) * Dc + dt0 + d] = acc[ti];
  }
}

// ---------------- depthwise K=3 conv on ctrl, + f16 copy ----------------
__global__ void hyena_shortconv(const float* __restrict__ proj,
                                const float* __restrict__ cw, const float* __restrict__ cb,
                                float* __restrict__ xc, _Float16* __restrict__ xch,
                                int Lc, int Dc, int stride, int coloff) {
  int gid = blockIdx.x * 256 + threadIdx.x;
  int d = gid % Dc;
  int m = gid / Dc;
  int t = m % Lc;
  int b = m / Lc;
  size_t base = (size_t)(b * Lc + t) * stride + coloff + d;
  float c0 = (t > 0)      ? proj[base - stride] : 0.0f;
  float c1 = proj[base];
  float c2 = (t < Lc - 1) ? proj[base + stride] : 0.0f;
  float val = cw[d] * c0 + cw[Dc + d] * c1 + cw[2 * Dc + d] * c2 + cb[d];
  size_t o = (size_t)m * Dc + d;
  xc[o]  = val;
  xch[o] = (_Float16)val;
}

// ---------------- WMMA f16 GEMM, TDM-fed double-buffered LDS ----------------
// out = A[MxK] @ WT^T + bias ; WT is [N][K] row-major.
// mode 0: out=acc+bias. mode 1: gate epilogue (vconv*xc*sigmoid), + f16 copy.
__global__ __launch_bounds__(256) void hyena_gemm(
    const _Float16* __restrict__ A, const _Float16* __restrict__ WT,
    const float* __restrict__ bias, float* __restrict__ out,
    int M, int N, int K, int mode,
    const float* __restrict__ mul1, const float* __restrict__ mul2,
    _Float16* __restrict__ outh) {
  __shared__ __align__(16) _Float16 As[2][128 * 32];
  __shared__ __align__(16) _Float16 Bs[2][64 * 32];
  const int tid  = threadIdx.x;
  const int lane = tid & 31;
  const int wid  = tid >> 5;
  const int wm = wid & 3, wn = wid >> 2;    // 4x2 wave grid, wave tile 32x32
  const int m0 = blockIdx.y * 128;
  const int n0 = blockIdx.x * 64;
  const int lh = lane >> 4;                 // K half select
  const int ll = lane & 15;
  const bool issuer = (wid == 0);

  const unsigned ldsA0 = (unsigned)(uintptr_t)&As[0][0];
  const unsigned ldsA1 = (unsigned)(uintptr_t)&As[1][0];
  const unsigned ldsB0 = (unsigned)(uintptr_t)&Bs[0][0];
  const unsigned ldsB1 = (unsigned)(uintptr_t)&Bs[1][0];

  v8f zero = {};
  v8f acc[2][2];
  acc[0][0] = zero; acc[0][1] = zero; acc[1][0] = zero; acc[1][1] = zero;

  // prologue: DMA first tile pair into buffer 0 (wave 0 only; TDM ignores EXEC)
  if (issuer) {
    tdm_load_2d_f16(A  + (size_t)m0 * K, ldsA0, 32u, 128u, (unsigned)K);
    tdm_load_2d_f16(WT + (size_t)n0 * K, ldsB0, 32u, 64u,  (unsigned)K);
  }
  asm volatile("" ::: "memory");

  int buf = 0;
  for (int kk = 0; kk < K; kk += 32, buf ^= 1) {
    if (issuer) {
      if (kk + 32 < K) {
        // prefetch next tile pair into the other buffer
        tdm_load_2d_f16(A  + (size_t)m0 * K + kk + 32, buf ? ldsA0 : ldsA1, 32u, 128u, (unsigned)K);
        tdm_load_2d_f16(WT + (size_t)n0 * K + kk + 32, buf ? ldsB0 : ldsB1, 32u, 64u,  (unsigned)K);
        // TENSORcnt completes in-order: <=2 outstanding => current buffer done
        __builtin_amdgcn_s_wait_tensorcnt((short)2);
      } else {
        __builtin_amdgcn_s_wait_tensorcnt((short)0);
      }
    }
    asm volatile("" ::: "memory");
    __syncthreads();   // publish DMA'd LDS to all 8 waves

    const _Float16* Ab = &As[buf][0];
    const _Float16* Bb = &Bs[buf][0];
    FragH a[2], b[2];
    #pragma unroll
    for (int mt = 0; mt < 2; mt++) {
      int m = wm * 32 + mt * 16 + ll;
      a[mt].q[0] = *(const uint4*)&Ab[m * 32 + 8 * lh];        // K 0..7 (+8 half)
      a[mt].q[1] = *(const uint4*)&Ab[m * 32 + 16 + 8 * lh];   // K 16..23 (+8 half)
    }
    #pragma unroll
    for (int nt = 0; nt < 2; nt++) {
      int n = wn * 32 + nt * 16 + ll;
      b[nt].q[0] = *(const uint4*)&Bb[n * 32 + 8 * lh];        // K 0..7 (+8 half)
      b[nt].q[1] = *(const uint4*)&Bb[n * 32 + 16 + 8 * lh];   // K 16..23 (+8 half)
    }
    #pragma unroll
    for (int mt = 0; mt < 2; mt++)
      #pragma unroll
      for (int nt = 0; nt < 2; nt++)
        acc[mt][nt] = __builtin_amdgcn_wmma_f32_16x16x32_f16(
            false, a[mt].v, false, b[nt].v, (short)0, acc[mt][nt], false, false);
    __syncthreads();   // all waves done reading buf before next DMA overwrites it
  }

  // epilogue: C/D layout — VGPR r: M=r (+8 for lanes 16..31), N=lane%16
  #pragma unroll
  for (int mt = 0; mt < 2; mt++) {
    #pragma unroll
    for (int nt = 0; nt < 2; nt++) {
      int ng = n0 + wn * 32 + nt * 16 + ll;
      float bv = bias[ng];
      #pragma unroll
      for (int r = 0; r < 8; r++) {
        int mg = m0 + wm * 32 + mt * 16 + r + 8 * lh;
        size_t idx = (size_t)mg * N + ng;
        float val = acc[mt][nt][r] + bv;
        if (mode == 1) {
          float g = 1.0f / (1.0f + expf(-val));
          float o = mul1[idx] * mul2[idx] * g;
          out[idx]  = o;
          outh[idx] = (_Float16)o;
        } else {
          out[idx] = val;
        }
      }
    }
  }
}

extern "C" void kernel_launch(void* const* d_in, const int* in_sizes, int n_in,
                              void* d_out, int out_size, void* d_ws, size_t ws_size,
                              hipStream_t stream) {
  (void)in_sizes; (void)n_in; (void)out_size;
  const float* x          = (const float*)d_in[0];
  const float* in_proj_w  = (const float*)d_in[1];
  const float* in_proj_b  = (const float*)d_in[2];
  const float* out_proj_w = (const float*)d_in[3];
  const float* out_proj_b = (const float*)d_in[4];
  const float* freqs      = (const float*)d_in[5];
  const float* mlp_w1     = (const float*)d_in[6];
  const float* mlp_b1     = (const float*)d_in[7];
  const float* mlp_w2     = (const float*)d_in[8];
  const float* mlp_b2     = (const float*)d_in[9];
  const float* mlp_w3     = (const float*)d_in[10];
  const float* mlp_b3     = (const float*)d_in[11];
  const float* decay      = (const float*)d_in[12];
  const float* conv_w     = (const float*)d_in[13];
  const float* conv_b     = (const float*)d_in[14];
  const float* gate_w     = (const float*)d_in[15];
  const float* gate_b     = (const float*)d_in[16];
  float* outp = (float*)d_out;

  const int Bb = 2, L = 4096, D = 1024;
  const int M = Bb * L, N3 = 3 * D;

  char* ws = (char*)d_ws;
  size_t need = (size_t)M * N3 * 4 + (size_t)L * D * 4 + 3 * (size_t)M * D * 4 +
                3 * (size_t)M * D * 2 + (size_t)D * N3 * 2 + 3 * (size_t)D * D * 2;
  if (ws_size < need) return;

  float*    proj  = (float*)ws;      ws += (size_t)M * N3 * 4;
  float*    hbuf  = (float*)ws;      ws += (size_t)L * D * 4;
  float*    vcur  = (float*)ws;      ws += (size_t)M * D * 4;
  float*    vconv = (float*)ws;      ws += (size_t)M * D * 4;
  float*    xc    = (float*)ws;      ws += (size_t)M * D * 4;
  _Float16* xh    = (_Float16*)ws;   ws += (size_t)M * D * 2;
  _Float16* xch   = (_Float16*)ws;   ws += (size_t)M * D * 2;
  _Float16* vh    = (_Float16*)ws;   ws += (size_t)M * D * 2;
  _Float16* wph   = (_Float16*)ws;   ws += (size_t)D * N3 * 2;
  _Float16* wgh   = (_Float16*)ws;   ws += (size_t)2 * D * D * 2;
  _Float16* woh   = (_Float16*)ws;

  // precision staging
  cvt_f16<<<(M * D) / 256, 256, 0, stream>>>(x, xh, M * D);
  cvt_transpose<<<(D * N3) / 256, 256, 0, stream>>>(in_proj_w, wph, D, N3);
  cvt_transpose<<<(D * D) / 256, 256, 0, stream>>>(gate_w, wgh, D, D);
  cvt_transpose<<<(D * D) / 256, 256, 0, stream>>>(gate_w + (size_t)D * D, wgh + (size_t)D * D, D, D);
  cvt_transpose<<<(D * D) / 256, 256, 0, stream>>>(out_proj_w, woh, D, D);

  // proj = x @ in_proj_w + b
  hyena_gemm<<<dim3(N3 / 64, M / 128), 256, 0, stream>>>(
      xh, wph, in_proj_b, proj, M, N3, D, 0, nullptr, nullptr, nullptr);

  for (int i = 0; i < 2; i++) {
    hyena_filter<<<L, 256, 0, stream>>>(
        freqs + i * 3, mlp_w1 + i * 7 * 64, mlp_b1 + i * 64,
        mlp_w2 + i * 64 * 64, mlp_b2 + i * 64,
        mlp_w3 + (size_t)i * 64 * D, mlp_b3 + i * D,
        decay + i * D, hbuf, L, D);

    const float* vin = (i == 0) ? proj : vcur;
    int vstride      = (i == 0) ? N3 : D;
    hyena_longconv<<<dim3(D / DT, L / TC, Bb), 256, 0, stream>>>(
        vin, vstride, hbuf, vconv, L, D);

    hyena_shortconv<<<(M * D) / 256, 256, 0, stream>>>(
        proj, conv_w + (size_t)i * 3 * D, conv_b + i * D, xc, xch, L, D, N3, (i + 1) * D);

    // v = vconv * xc * sigmoid(xc @ gate_w + gate_b); also emit f16 copy
    hyena_gemm<<<dim3(D / 64, M / 128), 256, 0, stream>>>(
        xch, wgh + (size_t)i * D * D, gate_b + i * D, vcur, M, D, D, 1, vconv, xc, vh);
  }

  // out = v @ out_proj_w + b
  hyena_gemm<<<dim3(D / 64, M / 128), 256, 0, stream>>>(
      vh, woh, out_proj_b, outp, M, D, D, 0, nullptr, nullptr, nullptr);
}
